// RegionProposalNetwork_35381940584819
// MI455X (gfx1250) — matrix operand order repaired and checked by reading, not compiled, not run on hardware
//
#include <hip/hip_runtime.h>

typedef __bf16 bf16;
typedef __attribute__((ext_vector_type(16))) __bf16 bf16x16;
typedef __attribute__((ext_vector_type(8)))  float  f32x8;
typedef __attribute__((ext_vector_type(4)))  unsigned int u32x4;
typedef __attribute__((ext_vector_type(8)))  int i32x8;
typedef __attribute__((ext_vector_type(4)))  int i32x4;

#if __has_builtin(__builtin_amdgcn_tensor_load_to_lds) && __has_builtin(__builtin_amdgcn_s_wait_tensorcnt)
#define USE_TDM 1
#else
#define USE_TDM 0
#endif

#define T_TOTAL  159882
#define K_CAND   4507
#define K_PAD    8192
#define NW       71
#define ROWS_PAD 4544
#define BM 64
#define BN 256
#define BK 32
#define KTOT 2304
#define XFORM_CLIP 4.135166556742356f

__constant__ int   c_aoff[6]   = {0,120000,150000,157500,159375,159882};
__constant__ int   c_cnt[5]    = {120000,30000,7500,1875,507};
__constant__ int   c_cumk[6]   = {0,1000,2000,3000,4000,4507};
__constant__ int   c_W[5]      = {200,100,50,25,13};
__constant__ int   c_stride[5] = {4,8,16,32,64};
__constant__ float c_size[5]   = {32.f,64.f,128.f,256.f,512.f};
__constant__ float c_ratio[3]  = {0.5f,1.f,2.f};

// ---------------------------------------------------------------- utilities
__device__ inline unsigned int keyflip(float f) {
  unsigned int u = __float_as_uint(f);
  return (u & 0x80000000u) ? ~u : (u | 0x80000000u);
}

__device__ inline int lvl_of(int t) {
  return (t < 120000) ? 0 : (t < 150000) ? 1 : (t < 157500) ? 2 : (t < 159375) ? 3 : 4;
}

// deterministic block-wide exclusive scan of a 0/1 flag (wave32 ballot based)
__device__ inline int blockScanBool(int flag, int* warpTot, int* totalOut) {
  const int tid = threadIdx.x;
  const int lane = tid & 31, w = tid >> 5;
  const int nw = (blockDim.x + 31) >> 5;
  unsigned long long b = __ballot(flag);
  int p = __popcll(b & ((1ull << lane) - 1ull));
  if (lane == 0) warpTot[w] = __popcll(b);
  __syncthreads();
  if (tid < 32) {
    int v = (tid < nw) ? warpTot[tid] : 0;
    #pragma unroll
    for (int off = 1; off < 32; off <<= 1) {
      int o = __shfl_up(v, off);
      if (lane >= off) v += o;
    }
    warpTot[tid] = v;            // inclusive scan
  }
  __syncthreads();
  int base = (w == 0) ? 0 : warpTot[w - 1];
  *totalOut = warpTot[31];
  int res = base + p;
  __syncthreads();               // protect warpTot reuse
  return res;
}

// ---------------------------------------------------------------- kernel 0: weight repack OIHW fp32 -> [co][tap*256+ci] bf16
__global__ __launch_bounds__(256)
void rpn_repack_w(const float* __restrict__ cw, bf16* __restrict__ wb) {
  int i = blockIdx.x * 256 + threadIdx.x;
  if (i >= 256 * KTOT) return;
  int co = i / KTOT, k = i - co * KTOT;
  int tap = k >> 8, ci = k & 255;
  wb[i] = (bf16)cw[(co * 256 + ci) * 9 + tap];
}

// ---------------------------------------------------------------- kernel 1: fused 3x3 conv (WMMA implicit GEMM) + ReLU + 1x1 heads
union FragU { uint4 q[2]; bf16x16 v; };

__global__ __launch_bounds__(256)
void rpn_conv3x3_fused(const float* __restrict__ X, const bf16* __restrict__ Wb,
                       const float* __restrict__ conv_b,
                       const float* __restrict__ cls_w, const float* __restrict__ cls_b,
                       const float* __restrict__ bbox_w, const float* __restrict__ bbox_b,
                       float* __restrict__ obj, float* __restrict__ deltas,
                       int H, int W, int aoff)
{
  __shared__ __align__(16) bf16 Asm[BM][BK + 8];   // 80B rows, 16B aligned
  __shared__ __align__(16) bf16 Bsm[BN][BK + 8];   // 80B rows (TDM pads 16B per 64B row)
  __shared__ __align__(16) bf16 Tsm[BM][BN + 8];

  const int tid  = threadIdx.x;
  const int lane = tid & 31;
  const int wave = tid >> 5;            // 8 waves: 2(M) x 4(N)
  const int wm = wave & 1, wn = wave >> 1;
  const int half = lane >> 4, l16 = lane & 15;
  const int n = blockIdx.y;
  const int HW = H * W;
  const int tile0 = blockIdx.x * BM;

  const int am  = tid & 63;             // pixel row this thread loads
  const int ac0 = tid >> 6;             // ci_local base (0..3)
  const int px  = tile0 + am;
  const int py  = px / W;
  const int pxx = px - py * W;
  const bool mvalid = px < HW;

  const float* Xn = X + (size_t)n * 256 * HW;

#if USE_TDM
  // Tensor DMA descriptor (CDNA5 ISA): 2D tile 32x256 of 2B
  // elements out of a 2304x256 tensor (row stride 2304 elems), LDS-padded by
  // 4 DWORDs after every 16 DWORDs -> 80B LDS rows matching Bsm[][BK+8].
  u32x4 g0;
  i32x8 g1;
  i32x4 g2 = {0, 0, 0, 0}, g3 = {0, 0, 0, 0};
  const unsigned ldsB = (unsigned)(unsigned long long)(void*)&Bsm[0][0];
  const unsigned long long gbase = (unsigned long long)(const void*)Wb;
  g0.x = 1u;                                        // count=1, user mode, no gather
  g0.y = ldsB;                                      // lds_addr
  g0.z = 0u; g0.w = 0u;                             // global addr set per K-block
  g1[0] = (int)((1u << 16) |                        // data_size = 2 bytes
                (1u << 20) |                        // pad_enable
                (3u << 22) |                        // pad_interval = 16 DWORDs
                (3u << 25));                        // pad_amount   = 4 DWORDs
  g1[1] = (int)((2304u & 0xFFFFu) << 16);           // tensor_dim0[15:0]
  g1[2] = (int)((2304u >> 16) | ((256u & 0xFFFFu) << 16)); // dim0 hi | tensor_dim1 lo
  g1[3] = (int)((256u >> 16) | (32u << 16));        // dim1 hi | tile_dim0 = 32
  g1[4] = (int)256u;                                // tile_dim1 = 256, tile_dim2 = 0
  g1[5] = (int)2304u;                               // tensor_dim0_stride lo32
  g1[6] = 0;                                        // stride hi | dim1_stride lo
  g1[7] = 0;
#endif

  f32x8 acc[2][4] = {};

  for (int kb = 0; kb < KTOT / BK; ++kb) {
    const int tap = kb >> 3;
    const int ci0 = (kb & 7) * BK;
    const int dy = tap / 3 - 1, dx = tap % 3 - 1;
    const int sy = py + dy, sx = pxx + dx;
    const bool svalid = mvalid && ((unsigned)sy < (unsigned)H) && ((unsigned)sx < (unsigned)W);
    const float* src = Xn + ((size_t)ci0 + ac0) * HW + sy * W + sx;
    #pragma unroll
    for (int i = 0; i < 8; ++i) {
      float v = svalid ? src[(size_t)(4 * i) * HW] : 0.0f;
      Asm[am][ac0 + 4 * i] = (bf16)v;
    }
#if USE_TDM
    if (wave == 0) {                                 // TDM is wave-level; one DMA per block
      unsigned long long ga = gbase + (unsigned long long)kb * (BK * 2);
      g0.z = (unsigned)ga;
      g0.w = (unsigned)((ga >> 32) & 0x01FFFFFFull) | (2u << 30);  // addr[56:32] | type=2
#if __clang_major__ >= 23
      i32x8 g4 = {0, 0, 0, 0, 0, 0, 0, 0};
      __builtin_amdgcn_tensor_load_to_lds(g0, g1, g2, g3, g4, 0);
#else
      __builtin_amdgcn_tensor_load_to_lds(g0, g1, g2, g3, 0);
#endif
      __builtin_amdgcn_s_wait_tensorcnt(0);
    }
#else
    {
      const uint4* ws4 = (const uint4*)(Wb + (size_t)tid * KTOT + kb * BK);
      uint4 w0 = ws4[0], w1 = ws4[1], w2 = ws4[2], w3 = ws4[3];
      uint4* dst = (uint4*)&Bsm[tid][0];
      dst[0] = w0; dst[1] = w1; dst[2] = w2; dst[3] = w3;
    }
#endif
    if (kb + 1 < KTOT / BK) {            // CDNA5 global_prefetch_b8 for next K-block
      const int nt = (kb + 1) >> 3, nc = ((kb + 1) & 7) * BK;
      __builtin_prefetch(Xn + ((size_t)nc + ac0) * HW + (py + nt / 3 - 1) * W + (pxx + nt % 3 - 1), 0, 1);
    }
    __syncthreads();

    // A fragment (16x32 bf16, ISA 7.12.2): lane l16=M; half=0 -> K0..7,K16..23; half=1 -> K8..15,K24..31
    FragU af[2], bfr[4];
    #pragma unroll
    for (int mi = 0; mi < 2; ++mi) {
      const uint4* p = (const uint4*)&Asm[wm * 32 + mi * 16 + l16][0];
      af[mi].q[0] = p[half];
      af[mi].q[1] = p[2 + half];
    }
    // B fragment (32x16): lane l16=N; half=0 -> K0..7,K16..23; half=1 -> K8..15,K24..31
    #pragma unroll
    for (int ni = 0; ni < 4; ++ni) {
      const uint4* p = (const uint4*)&Bsm[wn * 64 + ni * 16 + l16][0];
      bfr[ni].q[0] = p[half];
      bfr[ni].q[1] = p[2 + half];
    }
    #pragma unroll
    for (int ni = 0; ni < 4; ++ni)
      #pragma unroll
      for (int mi = 0; mi < 2; ++mi)
        acc[mi][ni] = __builtin_amdgcn_wmma_f32_16x16x32_bf16(
            false, af[mi].v, false, bfr[ni].v, (short)0, acc[mi][ni], false, false);
    __syncthreads();
  }

  // bias + ReLU into LDS tile (C/D layout: VGPR r -> M = r + 8*half, lane l16 -> N)
  #pragma unroll
  for (int ni = 0; ni < 4; ++ni) {
    const int nb = wn * 64 + ni * 16 + l16;
    const float bias = conv_b[nb];
    #pragma unroll
    for (int mi = 0; mi < 2; ++mi) {
      #pragma unroll
      for (int r = 0; r < 8; ++r) {
        const int m = wm * 32 + mi * 16 + 8 * half + r;
        float v = acc[mi][ni][r] + bias;
        Tsm[m][nb] = (bf16)(v > 0.f ? v : 0.f);
      }
    }
  }
  __syncthreads();

  // fused 1x1 heads: 64 pixels x (3 cls + 12 bbox) outputs
  for (int tt = tid; tt < BM * 15; tt += 256) {
    const int o = tt % 15, m = tt / 15;
    const int pg = tile0 + m;
    if (pg >= HW) continue;
    const float* wv = (o < 3) ? (cls_w + o * 256) : (bbox_w + (o - 3) * 256);
    float s = (o < 3) ? cls_b[o] : bbox_b[o - 3];
    #pragma unroll 8
    for (int c = 0; c < 256; ++c) s += (float)Tsm[m][c] * wv[c];
    const int baseI = n * T_TOTAL + aoff + pg * 3;
    if (o < 3) obj[baseI + o] = s;
    else { const int a = (o - 3) >> 2, j = (o - 3) & 3; deltas[(size_t)(baseI + a) * 4 + j] = s; }
  }
}

// ---------------------------------------------------------------- kernel 2: anchor generation + box decode + clip
__global__ __launch_bounds__(256)
void rpn_decode(const float* __restrict__ deltas, float* __restrict__ boxes)
{
  int g = blockIdx.x * 256 + threadIdx.x;
  if (g >= 2 * T_TOTAL) return;
  int t = g % T_TOTAL;
  int l = lvl_of(t);
  int rel = t - c_aoff[l];
  int p = rel / 3, a = rel - 3 * p;
  int Wl = c_W[l];
  int hh = p / Wl, ww = p - hh * Wl;
  float size = c_size[l], ratio = c_ratio[a];
  float hr = sqrtf(ratio), wr = 1.0f / hr;
  float wsz = wr * size, hsz = hr * size;
  float x1b = rintf(-wsz * 0.5f), y1b = rintf(-hsz * 0.5f);
  float x2b = rintf( wsz * 0.5f), y2b = rintf( hsz * 0.5f);
  float ox = (float)(ww * c_stride[l]), oy = (float)(hh * c_stride[l]);
  float ax1 = ox + x1b, ay1 = oy + y1b, ax2 = ox + x2b, ay2 = oy + y2b;
  float aw = ax2 - ax1, ah = ay2 - ay1;
  float acx = ax1 + 0.5f * aw, acy = ay1 + 0.5f * ah;
  const float* d = deltas + (size_t)g * 4;
  float dx = d[0], dy = d[1];
  float dw = fminf(d[2], XFORM_CLIP), dh = fminf(d[3], XFORM_CLIP);
  float pcx = dx * aw + acx, pcy = dy * ah + acy;
  float pw = expf(dw) * aw, ph = expf(dh) * ah;
  float x1 = fminf(fmaxf(pcx - 0.5f * pw, 0.f), 800.f);
  float y1 = fminf(fmaxf(pcy - 0.5f * ph, 0.f), 800.f);
  float x2 = fminf(fmaxf(pcx + 0.5f * pw, 0.f), 800.f);
  float y2 = fminf(fmaxf(pcy + 0.5f * ph, 0.f), 800.f);
  float* b = boxes + (size_t)g * 4;
  b[0] = x1; b[1] = y1; b[2] = x2; b[3] = y2;
}

// ---------------------------------------------------------------- kernel 3: deterministic per-(image,level) radix-select top-k
__global__ __launch_bounds__(1024)
void rpn_topk(const float* __restrict__ obj, int* __restrict__ topk)
{
  __shared__ unsigned int hist[1024];
  __shared__ int warpTot[32];
  __shared__ int sh_b, sh_above;
  const int bi = blockIdx.x;
  const int n = bi / 5, l = bi % 5;
  const int tid = threadIdx.x;
  const int nEl = c_cnt[l];
  const int kwant = nEl < 1000 ? nEl : 1000;
  int* out = topk + (n * 5 + l) * 1000;
  for (int i = tid; i < 1000; i += 1024) out[i] = -1;
  const float* src = obj + (size_t)n * T_TOTAL + c_aoff[l];

  unsigned int prefix = 0, pmask = 0;
  int countAbove = 0;
  for (int pass = 0; pass < 3; ++pass) {
    const int shift = 22 - pass * 10;
    hist[tid] = 0;
    __syncthreads();
    for (int i = tid; i < nEl; i += 1024) {
      unsigned int key = keyflip(src[i]);
      if ((key & pmask) == prefix) atomicAdd(&hist[(key >> shift) & 1023u], 1u);
    }
    __syncthreads();
    if (tid == 0) {
      int need = kwant - countAbove;
      int cum = 0, bsel = 0;
      for (int b = 1023; b >= 0; --b) {
        int h = (int)hist[b];
        if (cum + h >= need) { bsel = b; break; }
        cum += h;
      }
      sh_b = bsel; sh_above = cum;
    }
    __syncthreads();
    countAbove += sh_above;
    prefix |= ((unsigned int)sh_b) << shift;
    pmask  |= (1023u << shift);
    __syncthreads();
  }
  const unsigned int thr = prefix >> 2;
  const int r = kwant - countAbove;

  int cA = 0, cE = 0;
  for (int base = 0; base < nEl; base += 1024) {
    int i = base + tid;
    int fA = 0, fE = 0;
    if (i < nEl) {
      unsigned int k30 = keyflip(src[i]) >> 2;
      fA = (k30 > thr);
      fE = (k30 == thr);
    }
    int totA, totE;
    int pA = blockScanBool(fA, warpTot, &totA);
    int pE = blockScanBool(fE, warpTot, &totE);
    if (fA) out[cA + pA] = c_aoff[l] + i;
    if (fE) { int s = cE + pE; if (s < r) out[countAbove + s] = c_aoff[l] + i; }
    cA += totA; cE += totE;
  }
}

// ---------------------------------------------------------------- kernel 4: gather candidates + bitonic sort (desc) per image
__global__ __launch_bounds__(1024)
void rpn_gather_sort(const float* __restrict__ obj, const float* __restrict__ boxes,
                     const int* __restrict__ topk, int* __restrict__ candA,
                     float* __restrict__ sboxes, float* __restrict__ sscores,
                     int* __restrict__ slvl)
{
  __shared__ float key[K_PAD];
  __shared__ unsigned short pid[K_PAD];
  const int n = blockIdx.x, tid = threadIdx.x;
  const float NEG = -__builtin_inff();

  for (int j = tid; j < K_PAD; j += 1024) {
    float sc = NEG; int anchor = -1;
    if (j < K_CAND) {
      int l = (j < 1000) ? 0 : (j < 2000) ? 1 : (j < 3000) ? 2 : (j < 4000) ? 3 : 4;
      int slot = topk[(n * 5 + l) * 1000 + (j - c_cumk[l])];
      if (slot >= 0) {
        anchor = slot;
        const float* b = boxes + ((size_t)n * T_TOTAL + slot) * 4;
        if ((b[2] - b[0] >= 1e-3f) && (b[3] - b[1] >= 1e-3f)) {
          float o = obj[(size_t)n * T_TOTAL + slot];
          sc = 1.0f / (1.0f + expf(-o));
        }
      }
    }
    candA[n * K_PAD + j] = anchor;
    key[j] = sc; pid[j] = (unsigned short)j;
  }
  __syncthreads();

  for (int size = 2; size <= K_PAD; size <<= 1) {
    for (int stride = size >> 1; stride > 0; stride >>= 1) {
      for (int t = tid; t < K_PAD / 2; t += 1024) {
        int low = t & (stride - 1);
        int i = ((t - low) << 1) + low;
        int j = i + stride;
        bool desc = ((i & size) == 0);
        float ki = key[i], kj = key[j];
        bool sw = desc ? (ki < kj) : (ki > kj);
        if (sw) {
          key[i] = kj; key[j] = ki;
          unsigned short tp = pid[i]; pid[i] = pid[j]; pid[j] = tp;
        }
      }
      __syncthreads();
    }
  }

  for (int j = tid; j < K_CAND; j += 1024) {
    int anchor = candA[n * K_PAD + pid[j]];
    float sc = key[j];
    float4 bx = make_float4(0.f, 0.f, 0.f, 0.f);
    int l = -1;
    if (anchor >= 0 && sc != NEG) {
      const float* b = boxes + ((size_t)n * T_TOTAL + anchor) * 4;
      bx = make_float4(b[0], b[1], b[2], b[3]);
      l = lvl_of(anchor);
    }
    ((float4*)sboxes)[n * K_PAD + j] = bx;
    sscores[n * K_PAD + j] = sc;
    slvl[n * K_PAD + j] = l;
  }
}

// ---------------------------------------------------------------- kernel 5: NMS bit-mask matrix (same-level only == per-level offsets)
__global__ __launch_bounds__(64)
void rpn_nms_masks(const float* __restrict__ sboxes, const int* __restrict__ slvl,
                   unsigned long long* __restrict__ masks)
{
  __shared__ float4 cbx[64];
  __shared__ int    clv[64];
  const int n = blockIdx.z, rb = blockIdx.y, cb = blockIdx.x, tid = threadIdx.x;
  int j = cb * 64 + tid;
  if (j < K_CAND) {
    cbx[tid] = ((const float4*)sboxes)[n * K_PAD + j];
    clv[tid] = slvl[n * K_PAD + j];
  } else {
    cbx[tid] = make_float4(0.f, 0.f, 0.f, 0.f);
    clv[tid] = -2;
  }
  __syncthreads();
  int i = rb * 64 + tid;
  if (i >= K_CAND) return;
  float4 bi = ((const float4*)sboxes)[n * K_PAD + i];
  int li = slvl[n * K_PAD + i];
  float areai = (bi.z - bi.x) * (bi.w - bi.y);
  unsigned long long bits = 0ull;
  int jmax = K_CAND - cb * 64; if (jmax > 64) jmax = 64;
  for (int c = 0; c < jmax; ++c) {
    if (clv[c] != li) continue;
    float4 bj = cbx[c];
    float xx1 = fmaxf(bi.x, bj.x), yy1 = fmaxf(bi.y, bj.y);
    float xx2 = fminf(bi.z, bj.z), yy2 = fminf(bi.w, bj.w);
    float w = fmaxf(xx2 - xx1, 0.f), h = fmaxf(yy2 - yy1, 0.f);
    float inter = w * h;
    float areaj = (bj.z - bj.x) * (bj.w - bj.y);
    float iou = inter / (areai + areaj - inter);
    if (iou > 0.7f) bits |= (1ull << c);
  }
  masks[((size_t)n * ROWS_PAD + i) * NW + cb] = bits;
}

// ---------------------------------------------------------------- kernel 6: sequential mask reduce + compact output
__global__ __launch_bounds__(128)
void rpn_nms_reduce_out(const float* __restrict__ sboxes, const float* __restrict__ sscores,
                        const unsigned long long* __restrict__ masks, float* __restrict__ out)
{
  __shared__ unsigned long long rem[NW];
  __shared__ int warpTot[32];
  const int n = blockIdx.x, tid = threadIdx.x;
  const float NEG = -__builtin_inff();
  for (int c = tid; c < NW; c += 128) rem[c] = 0ull;
  __syncthreads();

  for (int i = 0; i < K_CAND; ++i) {
    bool kept = ((rem[i >> 6] >> (i & 63)) & 1ull) == 0ull;   // bit i never written by iter i
    if (kept) {
      const unsigned long long* mrow = masks + ((size_t)n * ROWS_PAD + i) * NW;
      int cb0 = i >> 6, low = i & 63;
      for (int c = cb0 + tid; c < NW; c += 128) {
        unsigned long long m = mrow[c];
        if (c == cb0) {
          unsigned long long lowmask = (low == 63) ? ~0ull : ((1ull << (low + 1)) - 1ull);
          m &= ~lowmask;                                       // only suppress j > i
        }
        if (m) rem[c] |= m;
      }
    }
    __syncthreads();
  }

  int cnt = 0;
  for (int base = 0; base < K_CAND; base += 128) {
    int i = base + tid;
    int flag = 0; float sc = NEG;
    if (i < K_CAND) {
      sc = sscores[n * K_PAD + i];
      bool sup = ((rem[i >> 6] >> (i & 63)) & 1ull) != 0ull;
      flag = (!sup && sc != NEG) ? 1 : 0;
    }
    int tot;
    int pos = blockScanBool(flag, warpTot, &tot);
    if (flag) {
      int slot = cnt + pos;
      if (slot < 1000) {
        const float* b = sboxes + ((size_t)n * K_PAD + i) * 4;
        float* o = out + ((size_t)n * 1000 + slot) * 5;
        o[0] = b[0]; o[1] = b[1]; o[2] = b[2]; o[3] = b[3]; o[4] = sc;
      }
    }
    cnt += tot;
  }
  int start = cnt < 1000 ? cnt : 1000;
  for (int s = start + tid; s < 1000; s += 128) {
    float* o = out + ((size_t)n * 1000 + s) * 5;
    o[0] = 0.f; o[1] = 0.f; o[2] = 0.f; o[3] = 0.f; o[4] = -1.0f;
  }
}

// ---------------------------------------------------------------- host
extern "C" void kernel_launch(void* const* d_in, const int* in_sizes, int n_in,
                              void* d_out, int out_size, void* d_ws, size_t ws_size,
                              hipStream_t stream) {
  (void)in_sizes; (void)n_in; (void)out_size; (void)ws_size;
  const float* feats[5];
  for (int i = 0; i < 5; ++i) feats[i] = (const float*)d_in[i];
  const float* conv_w = (const float*)d_in[5];
  const float* conv_b = (const float*)d_in[6];
  const float* cls_w  = (const float*)d_in[7];
  const float* cls_b  = (const float*)d_in[8];
  const float* bbox_w = (const float*)d_in[9];
  const float* bbox_b = (const float*)d_in[10];

  char* ws = (char*)d_ws;
  size_t off = 0;
  auto alloc = [&](size_t bytes) { size_t r = off; off += (bytes + 255) & ~(size_t)255; return r; };

  bf16*  wb      = (bf16*)(ws + alloc((size_t)KTOT * 256 * 2));
  float* obj     = (float*)(ws + alloc((size_t)2 * T_TOTAL * 4));
  float* deltas  = (float*)(ws + alloc((size_t)2 * T_TOTAL * 16));
  float* boxes   = (float*)(ws + alloc((size_t)2 * T_TOTAL * 16));
  int*   topk    = (int*)(ws + alloc((size_t)2 * 5 * 1000 * 4));
  int*   candA   = (int*)(ws + alloc((size_t)2 * K_PAD * 4));
  float* sboxes  = (float*)(ws + alloc((size_t)2 * K_PAD * 16));
  float* sscores = (float*)(ws + alloc((size_t)2 * K_PAD * 4));
  int*   slvl    = (int*)(ws + alloc((size_t)2 * K_PAD * 4));
  unsigned long long* masks = (unsigned long long*)(ws + alloc((size_t)2 * ROWS_PAD * NW * 8));

  rpn_repack_w<<<dim3((256 * KTOT + 255) / 256), 256, 0, stream>>>(conv_w, wb);

  struct Lv { int H, W, aoff, tiles; };
  const Lv lv[5] = {{200, 200, 0, 625}, {100, 100, 120000, 157}, {50, 50, 150000, 40},
                    {25, 25, 157500, 10}, {13, 13, 159375, 3}};
  for (int l = 0; l < 5; ++l) {
    rpn_conv3x3_fused<<<dim3(lv[l].tiles, 2), 256, 0, stream>>>(
        feats[l], wb, conv_b, cls_w, cls_b, bbox_w, bbox_b,
        obj, deltas, lv[l].H, lv[l].W, lv[l].aoff);
  }

  rpn_decode<<<dim3((2 * T_TOTAL + 255) / 256), 256, 0, stream>>>(deltas, boxes);
  rpn_topk<<<dim3(10), 1024, 0, stream>>>(obj, topk);
  rpn_gather_sort<<<dim3(2), 1024, 0, stream>>>(obj, boxes, topk, candA, sboxes, sscores, slvl);
  rpn_nms_masks<<<dim3(NW, NW, 2), 64, 0, stream>>>(sboxes, slvl, masks);
  rpn_nms_reduce_out<<<dim3(2), 128, 0, stream>>>(sboxes, sscores, masks, (float*)d_out);
}
